// GATClassifier_34239479284113
// MI455X (gfx1250) — compile-verified, matrix operand order checked by reading, not run on hardware
//
#include <hip/hip_runtime.h>
#include <hip/hip_bf16.h>

typedef __attribute__((ext_vector_type(16))) __bf16 v16bf;
typedef __attribute__((ext_vector_type(8)))  float  v8f;

#define NEG_SLOPE 0.2f

// ---------- helpers ----------

__device__ __forceinline__ unsigned short f2bf(float f) {
    unsigned u = __float_as_uint(f);
    unsigned r = u + 0x7FFFu + ((u >> 16) & 1u);   // round-to-nearest-even
    return (unsigned short)(r >> 16);
}

// order-preserving float->uint encoding for atomicMax on floats
__device__ __forceinline__ unsigned fenc(float f) {
    unsigned u = __float_as_uint(f);
    return (u & 0x80000000u) ? ~u : (u | 0x80000000u);
}
__device__ __forceinline__ float fdec(unsigned u) {
    return (u & 0x80000000u) ? __uint_as_float(u & 0x7FFFFFFFu)
                             : __uint_as_float(~u);
}

__device__ __forceinline__ void get_edge(const int* __restrict__ ei,
                                         long long n_edges, long long idx,
                                         int& s, int& d) {
    if (idx < n_edges) { s = ei[idx]; d = ei[n_edges + idx]; }
    else               { s = d = (int)(idx - n_edges); }     // self loops
}

// ---------- conversion / init kernels ----------

__global__ void k_f32_to_bf16(const float* __restrict__ in,
                              unsigned short* __restrict__ out, long long n) {
    long long i = (long long)blockIdx.x * blockDim.x + threadIdx.x;
    if (i < n) out[i] = f2bf(in[i]);
}

// W [K, Nc] f32 -> Wt [Nc, K] bf16
__global__ void k_transpose_bf16(const float* __restrict__ W,
                                 unsigned short* __restrict__ Wt,
                                 int K, int Nc) {
    int i = blockIdx.x * blockDim.x + threadIdx.x;
    if (i >= K * Nc) return;
    int n = i / K, k = i % K;
    Wt[(size_t)n * K + k] = f2bf(W[(size_t)k * Nc + n]);
}

__global__ void k_fill_u32(unsigned* __restrict__ p, unsigned v, long long n) {
    long long i = (long long)blockIdx.x * blockDim.x + threadIdx.x;
    if (i < n) p[i] = v;
}

// out[i] = bias[i % C]  (seeds output with bias before scatter-add)
__global__ void k_fill_bias(float* __restrict__ out, const float* __restrict__ bias,
                            int C, long long n) {
    long long i = (long long)blockIdx.x * blockDim.x + threadIdx.x;
    if (i < n) out[i] = bias[i % C];
}

// hbf[i] = bf16(relu(in[i] + bias[i % C]))
__global__ void k_bias_relu_bf16(const float* __restrict__ in,
                                 const float* __restrict__ bias,
                                 unsigned short* __restrict__ hbf,
                                 int C, long long n) {
    long long i = (long long)blockIdx.x * blockDim.x + threadIdx.x;
    if (i >= n) return;
    float v = in[i] + bias[i % C];
    hbf[i] = f2bf(v > 0.f ? v : 0.f);
}

// ---------- WMMA GEMM:  C[M,Nc] = A[M,128] * Bt[Nc,128]^T  (bf16 in, f32 out) ----------
// One wave (32 threads) computes one 16x16 tile; K=128 in 4 steps of 32.
// 16-bit A-fragment layout (ISA 7.12.2): lane L holds row M=L&15, half=L>>4;
// VGPR v holds K pair at k=(v<4?0:16)+8*half+2*(v&3). B fragment mirrors with
// column n = L&15 (Bt row-major makes its loads identical to A's).
__global__ void k_gemm_wmma_bf16(const unsigned short* __restrict__ A,
                                 const unsigned short* __restrict__ Bt,
                                 float* __restrict__ C, int Nc) {
    const int lane = threadIdx.x & 31;
    const int tileM = blockIdx.x;      // M/16 tiles
    const int tileN = blockIdx.y;      // Nc/16 tiles
    const int r15  = lane & 15;
    const int half = lane >> 4;

    const unsigned* arow = (const unsigned*)(A + (size_t)(tileM * 16 + r15) * 128);
    const unsigned* brow = (const unsigned*)(Bt + (size_t)(tileN * 16 + r15) * 128);

    v8f acc = {};
#pragma unroll
    for (int K0 = 0; K0 < 128; K0 += 32) {
        union { unsigned u[8]; v16bf v; } a, b;
#pragma unroll
        for (int v = 0; v < 8; ++v) {
            int k = ((v < 4) ? 0 : 16) + half * 8 + (v & 3) * 2;
            a.u[v] = arow[(K0 + k) >> 1];
            b.u[v] = brow[(K0 + k) >> 1];
        }
        acc = __builtin_amdgcn_wmma_f32_16x16x32_bf16(
                false, a.v, false, b.v, (short)0, acc, false, false);
    }

    // C/D layout: VGPR r -> row = r + 8*half, col = lane&15
#pragma unroll
    for (int r = 0; r < 8; ++r) {
        int row = tileM * 16 + r + 8 * half;
        int col = tileN * 16 + r15;
        C[(size_t)row * Nc + col] = acc[r];
    }
}

// ---------- attention coefficients:  as[n,h] = <h[n,h,:], a_s[h,:]> ----------
__global__ void k_alpha(const float* __restrict__ h,
                        const float* __restrict__ a_s, const float* __restrict__ a_d,
                        float* __restrict__ als, float* __restrict__ ald,
                        int n_nodes, int heads, int ch) {
    int i = blockIdx.x * blockDim.x + threadIdx.x;
    if (i >= n_nodes * heads) return;
    int n = i / heads, hd = i % heads;
    const float* row = h + (size_t)n * heads * ch + (size_t)hd * ch;
    const float* vs = a_s + (size_t)hd * ch;
    const float* vd = a_d + (size_t)hd * ch;
    float s = 0.f, d = 0.f;
    for (int c = 0; c < ch; ++c) { float x = row[c]; s += x * vs[c]; d += x * vd[c]; }
    als[i] = s; ald[i] = d;
}

// ---------- edge pass A: segment max of leaky_relu(as[src]+ad[dst]) ----------
__global__ void k_edge_max(const int* __restrict__ ei,
                           const float* __restrict__ als, const float* __restrict__ ald,
                           unsigned* __restrict__ m,
                           int heads, long long n_edges, long long total) {
    long long i = (long long)blockIdx.x * blockDim.x + threadIdx.x;
    if (i >= total * heads) return;
    long long e = i / heads; int hd = (int)(i % heads);
    int s, d; get_edge(ei, n_edges, e, s, d);
    float v = als[(size_t)s * heads + hd] + ald[(size_t)d * heads + hd];
    v = v > 0.f ? v : NEG_SLOPE * v;
    atomicMax(&m[(size_t)d * heads + hd], fenc(v));
}

// ---------- edge pass B: ex = exp(e - m[dst]); den[dst] += ex ----------
__global__ void k_edge_exp(const int* __restrict__ ei,
                           const float* __restrict__ als, const float* __restrict__ ald,
                           const unsigned* __restrict__ m,
                           float* __restrict__ exbuf, float* __restrict__ den,
                           int heads, long long n_edges, long long total) {
    long long i = (long long)blockIdx.x * blockDim.x + threadIdx.x;
    if (i >= total * heads) return;
    long long e = i / heads; int hd = (int)(i % heads);
    int s, d; get_edge(ei, n_edges, e, s, d);
    float v = als[(size_t)s * heads + hd] + ald[(size_t)d * heads + hd];
    v = v > 0.f ? v : NEG_SLOPE * v;
    float ex = __expf(v - fdec(m[(size_t)d * heads + hd]));
    exbuf[i] = ex;
    atomicAdd(&den[(size_t)d * heads + hd], ex);
}

// ---------- edge pass C: out[dst,c] += (ex/den[dst]) * h[src,c] ----------
__global__ void k_edge_aggr(const int* __restrict__ ei,
                            const float* __restrict__ exbuf, const float* __restrict__ den,
                            const float* __restrict__ h, float* __restrict__ out,
                            int heads, int ch, long long n_edges, long long total) {
    const int C = heads * ch;
    long long i = (long long)blockIdx.x * blockDim.x + threadIdx.x;
    if (i >= total * C) return;
    long long e = i / C; int c = (int)(i % C); int hd = c / ch;
    int s, d; get_edge(ei, n_edges, e, s, d);
    float alpha = exbuf[e * heads + hd] / (den[(size_t)d * heads + hd] + 1e-16f);
    atomicAdd(&out[(size_t)d * C + c], alpha * h[(size_t)s * C + c]);
}

// ---------- host orchestration ----------

extern "C" void kernel_launch(void* const* d_in, const int* in_sizes, int n_in,
                              void* d_out, int out_size, void* d_ws, size_t ws_size,
                              hipStream_t stream) {
    (void)n_in; (void)out_size; (void)ws_size; (void)in_sizes;
    constexpr long long N = 100000, E = 1600000, ET = E + N;
    constexpr int C1 = 128, HEADS = 4, HID = 32, C2 = 32;

    const float* x    = (const float*)d_in[0];
    const int*   ei   = (const int*)  d_in[1];
    const float* W1   = (const float*)d_in[2];
    const float* aS1  = (const float*)d_in[3];
    const float* aD1  = (const float*)d_in[4];
    const float* b1   = (const float*)d_in[5];
    const float* W2   = (const float*)d_in[6];
    const float* aS2  = (const float*)d_in[7];
    const float* aD2  = (const float*)d_in[8];
    const float* b2   = (const float*)d_in[9];
    float* out = (float*)d_out;

    // workspace layout (256B aligned), with cross-layer reuse
    char* ws = (char*)d_ws;
    size_t off = 0;
    auto alloc = [&](size_t bytes) { size_t o = off; off = (off + bytes + 255) & ~(size_t)255; return o; };
    unsigned short* xbf  = (unsigned short*)(ws + alloc(N * C1 * 2));   // also layer-2 bf16 input
    float*          h1   = (float*)(ws + alloc(N * C1 * 4));            // also h2 [N,32]
    float*          out1 = (float*)(ws + alloc(N * C1 * 4));
    float*          ex1  = (float*)(ws + alloc(ET * HEADS * 4));        // also ex2
    float*          as1  = (float*)(ws + alloc(N * HEADS * 4));        // also as2
    float*          ad1  = (float*)(ws + alloc(N * HEADS * 4));        // also ad2
    unsigned*       m1   = (unsigned*)(ws + alloc(N * HEADS * 4));      // also m2
    float*          den1 = (float*)(ws + alloc(N * HEADS * 4));        // also den2
    unsigned short* w1t  = (unsigned short*)(ws + alloc(128 * 128 * 2));
    unsigned short* w2t  = (unsigned short*)(ws + alloc(32 * 128 * 2));

    const int B = 256;
    auto blocks = [](long long n, int b) { return (unsigned)((n + b - 1) / b); };

    // ---- layer 1 ----
    k_f32_to_bf16<<<blocks(N * C1, B), B, 0, stream>>>(x, xbf, N * C1);
    k_transpose_bf16<<<blocks(128 * 128, B), B, 0, stream>>>(W1, w1t, 128, 128);
    k_transpose_bf16<<<blocks(128 * 32, B), B, 0, stream>>>(W2, w2t, 128, 32);

    k_gemm_wmma_bf16<<<dim3(N / 16, C1 / 16), 32, 0, stream>>>(xbf, w1t, h1, C1);

    k_alpha<<<blocks(N * HEADS, B), B, 0, stream>>>(h1, aS1, aD1, as1, ad1, N, HEADS, HID);

    k_fill_u32<<<blocks(N * HEADS, B), B, 0, stream>>>(m1, 0u, N * HEADS);
    k_fill_u32<<<blocks(N * HEADS, B), B, 0, stream>>>((unsigned*)den1, 0u, N * HEADS);
    k_fill_u32<<<blocks(N * C1, B), B, 0, stream>>>((unsigned*)out1, 0u, N * C1);

    k_edge_max <<<blocks(ET * HEADS, B), B, 0, stream>>>(ei, as1, ad1, m1, HEADS, E, ET);
    k_edge_exp <<<blocks(ET * HEADS, B), B, 0, stream>>>(ei, as1, ad1, m1, ex1, den1, HEADS, E, ET);
    k_edge_aggr<<<blocks(ET * C1, B), B, 0, stream>>>(ei, ex1, den1, h1, out1, HEADS, HID, E, ET);

    // h = relu(out1 + b1) -> bf16 (layer-2 GEMM input, reuses xbf)
    k_bias_relu_bf16<<<blocks(N * C1, B), B, 0, stream>>>(out1, b1, xbf, C1, N * C1);

    // ---- layer 2 (single head, mean over 1 head == identity) ----
    k_gemm_wmma_bf16<<<dim3(N / 16, C2 / 16), 32, 0, stream>>>(xbf, w2t, h1, C2);  // h2 in h1 buf

    k_alpha<<<blocks(N, B), B, 0, stream>>>(h1, aS2, aD2, as1, ad1, N, 1, C2);

    k_fill_u32<<<blocks(N, B), B, 0, stream>>>(m1, 0u, N);
    k_fill_u32<<<blocks(N, B), B, 0, stream>>>((unsigned*)den1, 0u, N);
    k_fill_bias<<<blocks(N * C2, B), B, 0, stream>>>(out, b2, C2, N * C2);  // seed with bias

    k_edge_max <<<blocks(ET, B), B, 0, stream>>>(ei, as1, ad1, m1, 1, E, ET);
    k_edge_exp <<<blocks(ET, B), B, 0, stream>>>(ei, as1, ad1, m1, ex1, den1, 1, E, ET);
    k_edge_aggr<<<blocks(ET * C2, B), B, 0, stream>>>(ei, ex1, den1, h1, out, 1, C2, E, ET);
}
